// GraphSAGEModel_26560077758613
// MI455X (gfx1250) — matrix operand order repaired and checked by reading, not compiled, not run on hardware
//
#include <hip/hip_runtime.h>
#include <hip/hip_bf16.h>
#include <math.h>

#define N_NODES 16384
#define NE      524288
#define IN_C    128
#define HID_C   128
#define OUT_C   64

typedef __attribute__((ext_vector_type(16))) __bf16 v16bf;
typedef __attribute__((ext_vector_type(8)))  float  v8f;
typedef __attribute__((ext_vector_type(2)))  float  v2f;

// ---------------------------------------------------------------- utilities
__global__ void zero_f32(float* __restrict__ p, int count) {
  int i = blockIdx.x * blockDim.x + threadIdx.x;
  if (i < count) p[i] = 0.0f;
}

// ------------------------------------------------------- edge scatter (mean)
// One thread per (edge, 4-channel chunk). Hardware fp32 atomics.
__global__ void scatter_accum(const float* __restrict__ feat,
                              const int* __restrict__ src,
                              const int* __restrict__ dst,
                              float* __restrict__ agg,
                              float* __restrict__ deg,
                              int do_deg) {
  int tid = blockIdx.x * blockDim.x + threadIdx.x;
  int e = tid >> 5;
  int q = tid & 31;               // float4 chunk: channels 4q..4q+3 (IN_C=128)
  if (e >= NE) return;
  int s = src[e];
  int d = dst[e];
  float4 v = ((const float4*)(feat + (size_t)s * IN_C))[q];
  float* ap = agg + (size_t)d * IN_C + q * 4;
  unsafeAtomicAdd(ap + 0, v.x);
  unsafeAtomicAdd(ap + 1, v.y);
  unsafeAtomicAdd(ap + 2, v.z);
  unsafeAtomicAdd(ap + 3, v.w);
  if (do_deg && q == 0) unsafeAtomicAdd(deg + d, 1.0f);
}

// ---------------- SAGE layer via f32 WMMA: out = mean@Wl + x@Wr + b  (K=128)
// Block = 256 threads = 8 waves; wave w computes a 16x16 output tile at
// rows [blockIdx.x*128 + w*16, +16), cols [blockIdx.y*16, +16).
// f32 16x16x4 fragment layout (ISA 7.12.2): lane half h -> A: V0=K(2h),
// V1=K(2h+1); B mirrors with lane = output column.
template <int M>
__global__ __launch_bounds__(256) void sage_wmma(
    const float* __restrict__ agg, const float* __restrict__ deg,
    const float* __restrict__ x,   const float* __restrict__ Wl,
    const float* __restrict__ Wr,  const float* __restrict__ b,
    float* __restrict__ out) {
  const int lane = threadIdx.x & 31;
  const int wave = threadIdx.x >> 5;
  const int m = lane & 15;                 // A row-in-tile / B col-in-tile
  const int h = lane >> 4;                 // lane half -> K sub-pair
  const int i0 = blockIdx.x * 128 + wave * 16;
  const int j0 = blockIdx.y * 16;
  const int row = i0 + m;

  const float inv = 1.0f / fmaxf(deg[row], 1.0f);
  const float* arow = agg + (size_t)row * 128;
  const float* xrow = x   + (size_t)row * 128;

  v8f c = {};
  for (int kb = 0; kb < 128; kb += 4) {
    const int k = kb + 2 * h;              // this lane covers K = k, k+1
    v2f am, ax, bl, br;
    am[0] = arow[k] * inv;  am[1] = arow[k + 1] * inv;
    ax[0] = xrow[k];        ax[1] = xrow[k + 1];
    bl[0] = Wl[k * M + j0 + m];       bl[1] = Wl[(k + 1) * M + j0 + m];
    br[0] = Wr[k * M + j0 + m];       br[1] = Wr[(k + 1) * M + j0 + m];
    c = __builtin_amdgcn_wmma_f32_16x16x4_f32(false, am, false, bl,
                                              (short)0, c, false, false);
    c = __builtin_amdgcn_wmma_f32_16x16x4_f32(false, ax, false, br,
                                              (short)0, c, false, false);
  }

  const float bias = b[j0 + m];
#pragma unroll
  for (int r = 0; r < 8; ++r) {
    int orow = i0 + r + 8 * h;
    out[(size_t)orow * M + j0 + m] = c[r] + bias;
  }
}

// ------------------------------------- row L2-normalize, emit f32 z + bf16 z
__global__ void normalize_rows(const float* __restrict__ h,
                               float* __restrict__ z_out,
                               __hip_bfloat16* __restrict__ zb) {
  int n = blockIdx.x * blockDim.x + threadIdx.x;
  if (n >= N_NODES) return;
  const float* r = h + (size_t)n * OUT_C;
  float s = 0.0f;
#pragma unroll 8
  for (int k = 0; k < OUT_C; ++k) { float v = r[k]; s += v * v; }
  float inv = 1.0f / fmaxf(sqrtf(s), 1e-12f);
#pragma unroll 8
  for (int k = 0; k < OUT_C; ++k) {
    float v = r[k] * inv;
    z_out[(size_t)n * OUT_C + k] = v;
    zb[(size_t)n * OUT_C + k] = __float2bfloat16(v);
  }
}

// ----------------------------------------------------- x_ = elu(z@W_lin + b)
__global__ void lin_elu(const float* __restrict__ z,
                        const float* __restrict__ W,
                        const float* __restrict__ b,
                        float* __restrict__ out) {
  int tid = blockIdx.x * blockDim.x + threadIdx.x;
  int n = tid / IN_C;
  int c = tid % IN_C;
  if (n >= N_NODES) return;
  const float* zr = z + (size_t)n * OUT_C;
  float acc = b[c];
#pragma unroll 8
  for (int k = 0; k < OUT_C; ++k) acc += zr[k] * W[k * IN_C + c];
  out[(size_t)n * IN_C + c] = acc > 0.0f ? acc : (__expf(acc) - 1.0f);
}

// --------------------------------------- A_pred = sigmoid(z z^T), bf16 WMMA
// Block = 256 threads = 8 waves; block tile = 128x128 output.
// Wave w owns rows [i0, i0+16); sweeps 8 column tiles of 16.
// K = 64 -> two v_wmma_f32_16x16x32_bf16 per tile with f32 accumulation.
// Epilogue: fast sigmoid (v_exp + v_rcp) and non-temporal streaming stores
// (A_pred is write-once; keep it out of L2 so z stays resident).
__global__ __launch_bounds__(256) void adj_decode_wmma(
    const __hip_bfloat16* __restrict__ zb, float* __restrict__ out) {
  const unsigned int* z32 = (const unsigned int*)zb;  // row stride = 32 dwords
  const int lane = threadIdx.x & 31;
  const int wave = threadIdx.x >> 5;
  const int m = lane & 15;   // A: row-in-tile   B: col-in-tile
  const int h = lane >> 4;   // lane half selects K sub-block

  const int i0    = blockIdx.x * 128 + wave * 16;
  const int jbase = blockIdx.y * 128;

  union F { v16bf v; unsigned int u[8]; };

  // A fragments (rows i0+m): v<4 -> K = 8h+2v ; v>=4 -> K = 16+8h+2(v-4)
  F a0, a1;
  const unsigned int* arow = z32 + (size_t)(i0 + m) * 32;
#pragma unroll
  for (int v = 0; v < 8; ++v) {
    int k = (v < 4) ? (h * 8 + 2 * v) : (16 + h * 8 + 2 * (v - 4));
    a0.u[v] = arow[k >> 1];           // K block 0..31
    a1.u[v] = arow[(k + 32) >> 1];    // K block 32..63
  }

  for (int jt = 0; jt < 8; ++jt) {
    const int j0 = jbase + jt * 16;
    // B fragments (cols j0+m): v<4 -> K = 8h+2v ; v>=4 -> K = 16+8h+2(v-4)
    F b0, b1;
    const unsigned int* brow = z32 + (size_t)(j0 + m) * 32;
#pragma unroll
    for (int v = 0; v < 8; ++v) {
      int k = (v < 4) ? (h * 8 + 2 * v) : (16 + h * 8 + 2 * (v - 4));
      b0.u[v] = brow[k >> 1];
      b1.u[v] = brow[(k + 32) >> 1];
    }

    v8f c = {};
    c = __builtin_amdgcn_wmma_f32_16x16x32_bf16(false, a0.v, false, b0.v,
                                                (short)0, c, false, false);
    c = __builtin_amdgcn_wmma_f32_16x16x32_bf16(false, a1.v, false, b1.v,
                                                (short)0, c, false, false);

    // D layout: VGPR r -> row (r + 8h), lane -> col m. Fused fast sigmoid.
#pragma unroll
    for (int r = 0; r < 8; ++r) {
      size_t row = (size_t)(i0 + r + 8 * h);
      float e  = __expf(-c[r]);
      float sg = __builtin_amdgcn_rcpf(1.0f + e);   // single v_rcp_f32
      __builtin_nontemporal_store(sg, &out[row * N_NODES + (j0 + m)]);
    }
  }
}

// --------------------------------------------------------------------- host
extern "C" void kernel_launch(void* const* d_in, const int* in_sizes, int n_in,
                              void* d_out, int out_size, void* d_ws, size_t ws_size,
                              hipStream_t stream) {
  const float* x     = (const float*)d_in[0];
  const float* W_l1  = (const float*)d_in[1];
  const float* W_r1  = (const float*)d_in[2];
  const float* b1    = (const float*)d_in[3];
  const float* W_l2  = (const float*)d_in[4];
  const float* W_r2  = (const float*)d_in[5];
  const float* b2    = (const float*)d_in[6];
  const float* W_lin = (const float*)d_in[7];
  const float* b_lin = (const float*)d_in[8];
  const int*   edges = (const int*)d_in[9];
  const int* src = edges;
  const int* dst = edges + NE;

  float* outA = (float*)d_out;                               // [N,N]
  float* outZ = outA + (size_t)N_NODES * N_NODES;            // [N,64]
  float* outX = outZ + (size_t)N_NODES * OUT_C;              // [N,128]

  char* ws = (char*)d_ws;
  float* agg = (float*)ws;  ws += (size_t)N_NODES * HID_C * sizeof(float);
  float* deg = (float*)ws;  ws += (size_t)N_NODES * sizeof(float);   // contiguous after agg
  float* h1  = (float*)ws;  ws += (size_t)N_NODES * HID_C * sizeof(float);
  float* h2  = (float*)ws;  ws += (size_t)N_NODES * OUT_C * sizeof(float);
  __hip_bfloat16* zbuf = (__hip_bfloat16*)ws;

  const int scatter_blocks = (NE * 32) / 256;   // 65536

  // ---- layer 1
  {
    int zc = N_NODES * HID_C + N_NODES;         // agg + deg in one sweep
    zero_f32<<<(zc + 255) / 256, 256, 0, stream>>>(agg, zc);
  }
  scatter_accum<<<scatter_blocks, 256, 0, stream>>>(x, src, dst, agg, deg, 1);
  sage_wmma<HID_C><<<dim3(N_NODES / 128, HID_C / 16), 256, 0, stream>>>(
      agg, deg, x, W_l1, W_r1, b1, h1);

  // ---- layer 2 (deg unchanged; re-zero agg only)
  zero_f32<<<(N_NODES * HID_C + 255) / 256, 256, 0, stream>>>(agg, N_NODES * HID_C);
  scatter_accum<<<scatter_blocks, 256, 0, stream>>>(h1, src, dst, agg, deg, 0);
  sage_wmma<OUT_C><<<dim3(N_NODES / 128, OUT_C / 16), 256, 0, stream>>>(
      agg, deg, h1, W_l2, W_r2, b2, h2);

  // ---- normalize -> z (f32 out) + bf16 copy for WMMA decode
  normalize_rows<<<N_NODES / 256, 256, 0, stream>>>(h2, outZ, zbuf);

  // ---- x_ = elu(z @ W_lin + b_lin)
  lin_elu<<<(N_NODES * IN_C) / 256, 256, 0, stream>>>(outZ, W_lin, b_lin, outX);

  // ---- A_pred = sigmoid(z z^T), bf16 WMMA, 128x128 tile per block
  dim3 grid(N_NODES / 128, N_NODES / 128);
  adj_decode_wmma<<<grid, 256, 0, stream>>>(zbuf, outA);
}
